// CausalSelfAttention_29068338659931
// MI455X (gfx1250) — matrix operand order repaired and checked by reading, not compiled, not run on hardware
//
#include <hip/hip_runtime.h>

// ---------------------------------------------------------------------------
// Causal self-attention forward for MI455X (gfx1250, wave32, WMMA).
//   out = softmax(mask(QK^T/sqrt(D))) V  with QKV = x@w_attn, out@w_proj
// B=4, T=2048, C=1024, H=16, D=64
// bf16 WMMA (f32 accumulate), flash-attention streaming softmax,
// async global->LDS K-tile copies, double-buffered K/V, transposed LDS
// layouts so every matrix-fragment read is a contiguous ds_load_b128.
// ---------------------------------------------------------------------------

typedef __attribute__((ext_vector_type(16))) __bf16 v16bf;
typedef __attribute__((ext_vector_type(8)))  float  v8f;

#define BDIM   4
#define TSEQ   2048
#define CDIM   1024
#define HHEADS 16
#define DHEAD  64
#define MTOT   (BDIM * TSEQ)   // 8192
#define N_QKV  (3 * CDIM)      // 3072

#define LDA 40   // padded leading dim for 32-wide K-slices (80B = 20 banks)
#define LDK 72   // padded leading dim for 64-wide tiles (144B = 36 banks)

// ----------------------------- fragment helpers ----------------------------
// A fragment: 16x32 bf16, row-major storage (ld), p0 at (row0, k0).
// lane m = lane&15, K contiguous 0-15 (+16 for upper half-wave).
__device__ __forceinline__ v16bf load_a_frag(const __bf16* p0, int ld) {
  const int lane = threadIdx.x & 31;
  const int m  = lane & 15;
  const int kb = (lane >> 4) << 4;
  const __bf16* p = p0 + m * ld + kb;
  v16bf a;
#pragma unroll
  for (int e = 0; e < 16; ++e) a[e] = p[e];
  return a;
}

// B fragment: logical B[k][n] (32x16) where storage is S[n][k] row-major (ld),
// p0 at S + col*ld + k0.  K-contiguous per lane -> 2x ds_load_b128.
__device__ __forceinline__ v16bf load_bT_frag(const __bf16* p0, int ld) {
  const int lane = threadIdx.x & 31;
  const int n  = lane & 15;
  const int kb = (lane >> 4) << 4;
  const __bf16* p = p0 + n * ld + kb;
  v16bf b;
#pragma unroll
  for (int e = 0; e < 16; ++e) b[e] = p[e];
  return b;
}

__device__ __forceinline__ v8f wmma_bf16(v16bf a, v16bf b, v8f c) {
  return __builtin_amdgcn_wmma_f32_16x16x32_bf16(
      false, a, false, b, (short)0, c, false, false);
}

__device__ __forceinline__ unsigned pack2(float x, float y) {
  unsigned short a = __builtin_bit_cast(unsigned short, (__bf16)x);
  unsigned short b = __builtin_bit_cast(unsigned short, (__bf16)y);
  return (unsigned)a | ((unsigned)b << 16);
}

// Async global->LDS 16B copy (GLOBAL_LOAD_ASYNC_TO_LDS_B128, ASYNCcnt).
// LDS destination address = low 32 bits of the generic shared pointer
// (aperture addressing, ISA 10.2).
__device__ __forceinline__ void async_copy_b128(void* lds_dst, const void* gsrc) {
  const unsigned lds_off = (unsigned)(size_t)lds_dst;
  asm volatile("global_load_async_to_lds_b128 %0, %1, off"
               :: "v"(lds_off), "v"((unsigned long long)(size_t)gsrc)
               : "memory");
}
__device__ __forceinline__ void wait_async0() {
  asm volatile("s_wait_asynccnt 0x0" ::: "memory");
}

// ---------------------------------------------------------------------------
// Kernel 1: QKV projection.  X[8192,1024] fp32 @ W[1024,3072] fp32,
// bf16 output scattered into head-major Q/K/V [B][H][T][D].
// 256 threads = 8 waves, 128x128 tile, K-step 32.
// ---------------------------------------------------------------------------
__global__ __launch_bounds__(256) void qkv_gemm_kernel(
    const float* __restrict__ X, const float* __restrict__ W,
    __bf16* __restrict__ Q, __bf16* __restrict__ K, __bf16* __restrict__ V) {
  __shared__ __bf16 Alds[128 * LDA];
  __shared__ __bf16 Btld[128 * LDA];   // B stored transposed: [n][k]
  const int tid  = threadIdx.x;
  const int wave = tid >> 5;
  const int wr   = wave >> 2;
  const int wc   = wave & 3;
  const int rb   = blockIdx.y * 128;
  const int cb   = blockIdx.x * 128;
  const int nloc = tid & 127;          // tile-local column for B fill
  const int kh   = (tid >> 7) << 4;    // 0 or 16

  const v8f vzero = {0.f, 0.f, 0.f, 0.f, 0.f, 0.f, 0.f, 0.f};
  v8f acc[4][2];
#pragma unroll
  for (int i = 0; i < 4; ++i)
#pragma unroll
    for (int j = 0; j < 2; ++j) acc[i][j] = vzero;

  for (int kk = 0; kk < CDIM; kk += 32) {
    __syncthreads();
    // A tile 128x32 fp32 -> bf16 (row-major, padded)
#pragma unroll
    for (int i = 0; i < 4; ++i) {
      const int lin = tid + i * 256;
      const int row = lin >> 3;
      const int g   = lin & 7;
      const float4 f = ((const float4*)(X + (size_t)(rb + row) * CDIM + kk))[g];
      ((uint2*)&Alds[row * LDA + (g << 2)])[0] =
          make_uint2(pack2(f.x, f.y), pack2(f.z, f.w));
    }
    // B tile 32x128 fp32 -> bf16, stored transposed [n][k]:
    // each thread owns one column half; global b32 loads stay coalesced.
    {
      float f[16];
#pragma unroll
      for (int j = 0; j < 16; ++j)
        f[j] = W[(size_t)(kk + kh + j) * N_QKV + cb + nloc];
#pragma unroll
      for (int j = 0; j < 4; ++j)
        ((uint2*)&Btld[nloc * LDA + kh + j * 4])[0] =
            make_uint2(pack2(f[4 * j], f[4 * j + 1]),
                       pack2(f[4 * j + 2], f[4 * j + 3]));
    }
    __syncthreads();
#pragma unroll
    for (int mt = 0; mt < 4; ++mt) {
      const v16bf a = load_a_frag(&Alds[(wr * 64 + mt * 16) * LDA], LDA);
#pragma unroll
      for (int nt = 0; nt < 2; ++nt) {
        const v16bf b = load_bT_frag(&Btld[(wc * 32 + nt * 16) * LDA], LDA);
        acc[mt][nt] = wmma_bf16(a, b, acc[mt][nt]);
      }
    }
  }

  const int lane = tid & 31;
  const int n    = lane & 15;
  const int half = lane >> 4;
#pragma unroll
  for (int mt = 0; mt < 4; ++mt) {
#pragma unroll
    for (int nt = 0; nt < 2; ++nt) {
#pragma unroll
      for (int r = 0; r < 8; ++r) {
        const int m_g = rb + wr * 64 + mt * 16 + r + half * 8;
        const int n_g = cb + wc * 32 + nt * 16 + n;
        const int which = n_g >> 10;
        const int c  = n_g & 1023;
        const int h  = c >> 6;
        const int d  = c & 63;
        const int b_ = m_g >> 11;
        const int t  = m_g & 2047;
        __bf16* dst = (which == 0) ? Q : (which == 1) ? K : V;
        dst[((((size_t)b_ * HHEADS + h) * TSEQ + t) << 6) + d] =
            (__bf16)acc[mt][nt][r];
      }
    }
  }
}

// ---------------------------------------------------------------------------
// Kernel 2: flash attention.
// Grid: (B*H, T/64), 128 threads = 4 waves, wave owns 16 query rows.
// Double-buffered K/V tiles: K async-copied row-major (consumed K-contiguous
// as B^T fragments), V transposed during fill so P.V fragments are also
// contiguous.  Online softmax over 16-lane half-wave row groups.
// ---------------------------------------------------------------------------
__global__ __launch_bounds__(128) void attn_kernel(
    const __bf16* __restrict__ Q, const __bf16* __restrict__ Kg,
    const __bf16* __restrict__ Vg, __bf16* __restrict__ O) {
  __shared__ __bf16 Klds[2][64 * LDK];   // [key][d]
  __shared__ __bf16 Vt[2][64 * LDK];     // [d][key]  (transposed)
  __shared__ __bf16 Plds[4 * 16 * 64];

  const int tid  = threadIdx.x;
  const int lane = tid & 31;
  const int wave = tid >> 5;
  const int bh   = blockIdx.x;
  const int qb   = blockIdx.y;
  const int qrow = qb * 64 + wave * 16;
  const size_t head_base = (size_t)bh * TSEQ * DHEAD;

  const v16bf qa0 = load_a_frag(Q + head_base + (size_t)qrow * DHEAD + 0, DHEAD);
  const v16bf qa1 = load_a_frag(Q + head_base + (size_t)qrow * DHEAD + 32, DHEAD);

  const v8f vzero = {0.f, 0.f, 0.f, 0.f, 0.f, 0.f, 0.f, 0.f};
  float m_i[8], l_i[8];
  v8f o_acc[4];
#pragma unroll
  for (int r = 0; r < 8; ++r) { m_i[r] = -1e30f; l_i[r] = 0.f; }
#pragma unroll
  for (int dt = 0; dt < 4; ++dt) o_acc[dt] = vzero;

  const int n    = lane & 15;
  const int half = lane >> 4;
  __bf16* Pw = &Plds[wave * 16 * 64];

  // tile prefetch: K via async b128 copies, V via load + transposed scatter
  auto prefetch = [&](int kt, int buf) {
    const __bf16* ksrc = Kg + head_base + (size_t)kt * 64 * DHEAD;
    const __bf16* vsrc = Vg + head_base + (size_t)kt * 64 * DHEAD;
#pragma unroll
    for (int i = 0; i < 4; ++i) {
      const int lin = tid + i * 128;     // 16B chunk, 0..511
      const int e   = lin << 3;
      const int row = e >> 6;            // key
      const int col = e & 63;            // d
      async_copy_b128(&Klds[buf][row * LDK + col], ksrc + e);
      const uint4 u = *(const uint4*)(vsrc + e);
      const __bf16* vb = (const __bf16*)&u;
#pragma unroll
      for (int j = 0; j < 8; ++j)
        Vt[buf][(col + j) * LDK + row] = vb[j];
    }
  };

  prefetch(0, 0);

  for (int kt = 0; kt <= qb; ++kt) {
    const int buf = kt & 1;
    wait_async0();                       // own K async copies landed
    __syncthreads();                     // publish tile kt to all waves
    if (kt < qb) prefetch(kt + 1, buf ^ 1);

    // S = (Q K^T) * 1/sqrt(D)
    v8f s[4];
#pragma unroll
    for (int nt = 0; nt < 4; ++nt) {
      s[nt] = vzero;
      s[nt] = wmma_bf16(qa0, load_bT_frag(&Klds[buf][(nt * 16) * LDK + 0], LDK), s[nt]);
      s[nt] = wmma_bf16(qa1, load_bT_frag(&Klds[buf][(nt * 16) * LDK + 32], LDK), s[nt]);
    }
    const float scale = 0.125f;
#pragma unroll
    for (int nt = 0; nt < 4; ++nt)
#pragma unroll
      for (int r = 0; r < 8; ++r) s[nt][r] *= scale;

    if (kt == qb) {                      // diagonal tile: causal mask
#pragma unroll
      for (int nt = 0; nt < 4; ++nt) {
        const int key = kt * 64 + nt * 16 + n;
#pragma unroll
        for (int r = 0; r < 8; ++r) {
          const int row = qrow + r + half * 8;
          if (key > row) s[nt][r] = -1e30f;
        }
      }
    }

    // online softmax; rows live across the 16 lanes of each half-wave
#pragma unroll
    for (int r = 0; r < 8; ++r) {
      float t = fmaxf(fmaxf(s[0][r], s[1][r]), fmaxf(s[2][r], s[3][r]));
#pragma unroll
      for (int off = 8; off >= 1; off >>= 1)
        t = fmaxf(t, __shfl_xor(t, off, 32));
      const float mnew  = fmaxf(m_i[r], t);
      const float alpha = __expf(m_i[r] - mnew);
      m_i[r] = mnew;
      l_i[r] *= alpha;
      float rsum = 0.f;
#pragma unroll
      for (int nt = 0; nt < 4; ++nt) {
        const float p = __expf(s[nt][r] - mnew);
        s[nt][r] = p;
        rsum += p;
      }
#pragma unroll
      for (int off = 8; off >= 1; off >>= 1)
        rsum += __shfl_xor(rsum, off, 32);
      l_i[r] += rsum;
#pragma unroll
      for (int dt = 0; dt < 4; ++dt) o_acc[dt][r] *= alpha;
    }

    // P (C layout) -> LDS -> A fragments
#pragma unroll
    for (int nt = 0; nt < 4; ++nt)
#pragma unroll
      for (int r = 0; r < 8; ++r)
        Pw[(r + half * 8) * 64 + nt * 16 + n] = (__bf16)s[nt][r];
    asm volatile("s_wait_dscnt 0" ::: "memory");

    const v16bf pa0 = load_a_frag(Pw + 0, 64);
    const v16bf pa1 = load_a_frag(Pw + 32, 64);
#pragma unroll
    for (int dt = 0; dt < 4; ++dt) {
      o_acc[dt] = wmma_bf16(pa0, load_bT_frag(&Vt[buf][(dt * 16) * LDK + 0], LDK), o_acc[dt]);
      o_acc[dt] = wmma_bf16(pa1, load_bT_frag(&Vt[buf][(dt * 16) * LDK + 32], LDK), o_acc[dt]);
    }
  }

  // write O merged back to [B][T][C] bf16
  const int b_ = bh >> 4;
  const int h  = bh & 15;
#pragma unroll
  for (int dt = 0; dt < 4; ++dt) {
#pragma unroll
    for (int r = 0; r < 8; ++r) {
      const int t = qrow + r + half * 8;
      const float o = o_acc[dt][r] / l_i[r];
      O[(((size_t)b_ * TSEQ + t) << 10) + h * 64 + dt * 16 + n] = (__bf16)o;
    }
  }
}

// ---------------------------------------------------------------------------
// Kernel 3: output projection.  A[8192,1024] bf16 @ W[1024,1024] fp32 -> fp32.
// ---------------------------------------------------------------------------
__global__ __launch_bounds__(256) void proj_gemm_kernel(
    const __bf16* __restrict__ A, const float* __restrict__ W,
    float* __restrict__ Out) {
  __shared__ __bf16 Alds[128 * LDA];
  __shared__ __bf16 Btld[128 * LDA];
  const int tid  = threadIdx.x;
  const int wave = tid >> 5;
  const int wr   = wave >> 2;
  const int wc   = wave & 3;
  const int rb   = blockIdx.y * 128;
  const int cb   = blockIdx.x * 128;
  const int nloc = tid & 127;
  const int kh   = (tid >> 7) << 4;

  const v8f vzero = {0.f, 0.f, 0.f, 0.f, 0.f, 0.f, 0.f, 0.f};
  v8f acc[4][2];
#pragma unroll
  for (int i = 0; i < 4; ++i)
#pragma unroll
    for (int j = 0; j < 2; ++j) acc[i][j] = vzero;

  for (int kk = 0; kk < CDIM; kk += 32) {
    __syncthreads();
    // A tile 128x32 (already bf16)
#pragma unroll
    for (int i = 0; i < 4; ++i) {
      const int lin = tid + i * 256;
      const int row = lin >> 3;
      const int g   = lin & 7;
      const uint2 u = ((const uint2*)(A + (size_t)(rb + row) * CDIM + kk))[g];
      ((uint2*)&Alds[row * LDA + (g << 2)])[0] = u;
    }
    // B tile 32x128 fp32 -> bf16 transposed [n][k]
    {
      float f[16];
#pragma unroll
      for (int j = 0; j < 16; ++j)
        f[j] = W[(size_t)(kk + kh + j) * CDIM + cb + nloc];
#pragma unroll
      for (int j = 0; j < 4; ++j)
        ((uint2*)&Btld[nloc * LDA + kh + j * 4])[0] =
            make_uint2(pack2(f[4 * j], f[4 * j + 1]),
                       pack2(f[4 * j + 2], f[4 * j + 3]));
    }
    __syncthreads();
#pragma unroll
    for (int mt = 0; mt < 4; ++mt) {
      const v16bf a = load_a_frag(&Alds[(wr * 64 + mt * 16) * LDA], LDA);
#pragma unroll
      for (int nt = 0; nt < 2; ++nt) {
        const v16bf b = load_bT_frag(&Btld[(wc * 32 + nt * 16) * LDA], LDA);
        acc[mt][nt] = wmma_bf16(a, b, acc[mt][nt]);
      }
    }
  }

  const int lane = tid & 31;
  const int n    = lane & 15;
  const int half = lane >> 4;
#pragma unroll
  for (int mt = 0; mt < 4; ++mt) {
#pragma unroll
    for (int nt = 0; nt < 2; ++nt) {
#pragma unroll
      for (int r = 0; r < 8; ++r) {
        const int m_g = rb + wr * 64 + mt * 16 + r + half * 8;
        const int n_g = cb + wc * 32 + nt * 16 + n;
        Out[(size_t)m_g * CDIM + n_g] = acc[mt][nt][r];
      }
    }
  }
}

// ---------------------------------------------------------------------------
extern "C" void kernel_launch(void* const* d_in, const int* in_sizes, int n_in,
                              void* d_out, int out_size, void* d_ws,
                              size_t ws_size, hipStream_t stream) {
  const float* x      = (const float*)d_in[0];
  const float* w_attn = (const float*)d_in[1];
  const float* w_proj = (const float*)d_in[2];
  float* out = (float*)d_out;

  const size_t per = (size_t)BDIM * TSEQ * CDIM;  // 8,388,608 elements
  __bf16* Qws = (__bf16*)d_ws;                    // [B][H][T][D]
  __bf16* Kws = Qws + per;
  __bf16* Vws = Kws + per;
  __bf16* Ows = Vws + per;                        // [B][T][C]
  // total workspace: 64 MB

  qkv_gemm_kernel<<<dim3(N_QKV / 128, MTOT / 128), 256, 0, stream>>>(
      x, w_attn, Qws, Kws, Vws);
  attn_kernel<<<dim3(BDIM * HHEADS, TSEQ / 64), 128, 0, stream>>>(
      Qws, Kws, Vws, Ows);
  proj_gemm_kernel<<<dim3(CDIM / 128, MTOT / 128), 256, 0, stream>>>(
      Ows, w_proj, out);
}
